// FullAttentionConv_10368051053158
// MI455X (gfx1250) — matrix operand order repaired and checked by reading, not compiled, not run on hardware
//
#include <hip/hip_runtime.h>
#include <hip/hip_bf16.h>

// ---------------- problem constants ----------------
#define C     512          // IN_CH == OUT_CH
#define H     8            // heads
#define HD    64           // head dim
#define TILE  16           // nodes per WMMA tile
#define KSTEPS (C / 32)    // 16 k-steps of 32 for the 512-deep projections

// ---------------- workspace layout (bytes) ----------------
#define WS_SUMSQ   0                         // [0]=sumsqQ [1]=sumsqK [2]=rnormQ
#define WS_KEYSUM  64                        // 512 f32 raw key_sum
#define WS_KTV     4096                      // 8*64*64 f32 raw ktv
#define WS_WH      (4096 + 131072)           // 3*512*512 f16 weights (Q,K,V)
#define WS_MH      (WS_WH + 3*C*C*2)         // 8*64*64 f16 (ktv-corr)*rnormK, [h][d][m]
#define WS_KSS     (WS_MH + H*HD*HD*2)       // 512 f32 corrected key_sum*rnormQ*rnormK

// ---------------- dynamic LDS partitioning (bytes) ----------------
// k1: [Xs32 f32 16x512 | Xs f16 16x520 | Kt 8x1024h | Vt 8x1024h | red 16f]
#define L1_XS32   0
#define L1_XS     32768
#define L1_KT     (32768 + 16640)
#define L1_VT     (L1_KT + 16384)
#define L1_RED    (L1_VT + 16384)
#define L1_BYTES  (L1_RED + 64)              // 82240
// k3: [Xs32 | Xs | Qs 8x1280h | den 8x16f]
#define L3_QS     (32768 + 16640)
#define L3_DEN    (L3_QS + 20480)
#define L3_BYTES  (L3_DEN + 512)             // 70400

typedef __attribute__((ext_vector_type(16))) _Float16 v16h;
typedef __attribute__((ext_vector_type(8)))  _Float16 v8h;
typedef __attribute__((ext_vector_type(8)))  float    v8f;
typedef unsigned int v4u __attribute__((ext_vector_type(4)));
typedef int          v4i __attribute__((ext_vector_type(4)));
typedef int          v8i __attribute__((ext_vector_type(8)));

#if defined(__has_builtin)
#if __has_builtin(__builtin_amdgcn_tensor_load_to_lds) && \
    __has_builtin(__builtin_amdgcn_s_wait_tensorcnt)
#define HAVE_TDM 1
#endif
#endif
#ifndef HAVE_TDM
#define HAVE_TDM 0
#endif

__device__ __forceinline__ v8f zero_v8f() {
  v8f r;
#pragma unroll
  for (int i = 0; i < 8; ++i) r[i] = 0.0f;
  return r;
}
__device__ __forceinline__ v8h zero_v8h() {
  v8h r;
#pragma unroll
  for (int i = 0; i < 8; ++i) r[i] = (_Float16)0.0f;
  return r;
}
__device__ __forceinline__ v16h pack16(v8h lo, v8h hi) {
  v16h r;
#pragma unroll
  for (int i = 0; i < 8; ++i) { r[i] = lo[i]; r[8 + i] = hi[i]; }
  return r;
}
__device__ __forceinline__ v8f wmma16(v16h a, v16h b, v8f c) {
  // D = A(16x32 f16) * B(32x16 f16) + C(16x16 f32)
  return __builtin_amdgcn_wmma_f32_16x16x32_f16(false, a, false, b,
                                                (short)0, c, false, false);
}

#if HAVE_TDM
// Issue one TDM DMA: 16x512 f32 tile (rows past rem_rows read as zero) -> LDS.
// D# per CDNA5 ISA ch.8: group0 = {count/flags, lds_addr, global_addr, type=2},
// group1 = {data_size=4B, tensor_dim0=512, tensor_dim1=rem, tile=512x16, stride0=512}.
__device__ __forceinline__ void tdm_load_x(const float* gaddr, int rem_rows,
                                           unsigned lds_off) {
  unsigned long long ga = (unsigned long long)(size_t)gaddr;
  v4u g0;
  g0[0] = 1u;                                         // count=1, user mode
  g0[1] = lds_off;                                    // lds_addr (bytes)
  g0[2] = (unsigned)ga;                               // global_addr[31:0]
  g0[3] = (unsigned)((ga >> 32) & 0x01ffffffu)        // global_addr[56:32]
        | (2u << 30);                                 // type=2 ("image")
  unsigned d1 = (unsigned)rem_rows;
  v8i g1;
  g1[0] = (int)(2u << 16);                            // data_size=2 (4 bytes)
  g1[1] = (int)((unsigned)C << 16);                   // tensor_dim0[15:0]
  g1[2] = (int)((d1 & 0xffffu) << 16);                // dim0[31:16]=0 | dim1[15:0]
  g1[3] = (int)((d1 >> 16) | ((unsigned)C << 16));    // dim1[31:16] | tile_dim0=512
  g1[4] = (int)TILE;                                  // tile_dim1=16, tile_dim2=0
  g1[5] = (int)C;                                     // tensor_dim0_stride[31:0]=512
  g1[6] = 0;
  g1[7] = 0;
  v4i z4 = {0, 0, 0, 0};
#if defined(__clang_major__) && (__clang_major__ >= 23)
  v8i z8 = {0, 0, 0, 0, 0, 0, 0, 0};
  __builtin_amdgcn_tensor_load_to_lds(g0, g1, z4, z4, z8, 0);
#else
  __builtin_amdgcn_tensor_load_to_lds(g0, g1, z4, z4, 0);
#endif
}
#endif

// =====================================================================
// K0a: zero the reduction scratch (sumsq, key_sum, ktv)  -> 33792 f32
// =====================================================================
__global__ void zero_red(float* __restrict__ wsf) {
  int i = blockIdx.x * 256 + threadIdx.x;   // grid = 132 blocks -> 33792 exact
  wsf[i] = 0.0f;
}

// =====================================================================
// K0b: convert Wq|Wk|Wv (f32 row-major [o][k]) -> f16, concatenated
// =====================================================================
__global__ void cvt_w(const float* __restrict__ wq, const float* __restrict__ wk,
                      const float* __restrict__ wv, _Float16* __restrict__ wh) {
  int i = blockIdx.x * 256 + threadIdx.x;   // grid = 3072 -> 786432 exact
  if (i >= 3 * C * C) return;
  int m = i >> 18;
  int r = i & (C * C - 1);
  const float* s = (m == 0) ? wq : (m == 1) ? wk : wv;
  wh[i] = (_Float16)s[r];
}

// =====================================================================
// K1: projections + global reductions (no per-element tail masking:
//     padded rows contribute exact bias terms, corrected in K2).
//     X tiles arrive via TDM (double-buffered against compute).
// =====================================================================
__global__ __launch_bounds__(256) void
k1_reduce(const float* __restrict__ x,
          const float* __restrict__ bq, const float* __restrict__ bk,
          const float* __restrict__ bv,
          float* __restrict__ wsf, const _Float16* __restrict__ wh,
          int n, int ntiles) {
  extern __shared__ char smem[];
  float*    Xs32 = (float*)(smem + L1_XS32);
  _Float16* Xs   = (_Float16*)(smem + L1_XS);
  _Float16* Kt   = (_Float16*)(smem + L1_KT);
  _Float16* Vt   = (_Float16*)(smem + L1_VT);
  float*    red  = (float*)(smem + L1_RED);

  const int w    = threadIdx.x >> 5;        // wave id == head id
  const int lane = threadIdx.x & 31;
  const int half = lane >> 4;
  const int lc   = lane & 15;
  const int h    = w;

  v8f ktv_acc[16];
#pragma unroll
  for (int i = 0; i < 16; ++i) ktv_acc[i] = zero_v8f();
  float ks_acc[4] = {0.f, 0.f, 0.f, 0.f};
  float ssq = 0.f, ssk = 0.f;

#if HAVE_TDM
  if (w == 0)                               // one DMA per block (EXEC ignored)
    tdm_load_x(x + (size_t)blockIdx.x * TILE * C, n - blockIdx.x * TILE,
               (unsigned)(unsigned long long)(size_t)(void*)Xs32);
#endif

  for (int t = blockIdx.x; t < ntiles; t += gridDim.x) {
    const int nb = t * TILE;
#if HAVE_TDM
    if (w == 0) __builtin_amdgcn_s_wait_tensorcnt(0);
    __syncthreads();                        // tile t resident & visible
    // f32 LDS -> f16 fragment-layout LDS
    for (int i = threadIdx.x; i < TILE * C / 4; i += 256) {
      int row = i >> 7;
      int c4  = (i & 127) * 4;
      float4 xv = *(const float4*)&Xs32[row * C + c4];
      Xs[row * 520 + c4 + 0] = (_Float16)xv.x;
      Xs[row * 520 + c4 + 1] = (_Float16)xv.y;
      Xs[row * 520 + c4 + 2] = (_Float16)xv.z;
      Xs[row * 520 + c4 + 3] = (_Float16)xv.w;
    }
    __syncthreads();                        // Xs ready; Xs32 free
    {
      int tn = t + gridDim.x;
      if (w == 0 && tn < ntiles)            // DMA tile t+gridDim.x behind the WMMAs
        tdm_load_x(x + (size_t)tn * TILE * C, n - tn * TILE,
                   (unsigned)(unsigned long long)(size_t)(void*)Xs32);
    }
#else
    for (int i = threadIdx.x; i < TILE * C / 4; i += 256) {
      int row = i >> 7;
      int c4  = (i & 127) * 4;
      int node = nb + row;
      float4 xv = make_float4(0.f, 0.f, 0.f, 0.f);
      if (node < n) xv = *(const float4*)(x + (size_t)node * C + c4);
      Xs[row * 520 + c4 + 0] = (_Float16)xv.x;
      Xs[row * 520 + c4 + 1] = (_Float16)xv.y;
      Xs[row * 520 + c4 + 2] = (_Float16)xv.z;
      Xs[row * 520 + c4 + 3] = (_Float16)xv.w;
    }
    int tn = t + gridDim.x;
    if (tn < ntiles && (tn * TILE + (int)(threadIdx.x >> 4)) < n)
      __builtin_prefetch(x + (size_t)tn * TILE * C + threadIdx.x * 32, 0, 0);
    __syncthreads();
#endif

    // ---- K_h : GEMM + key_sum/sumsq folds + stage K^T ----
#pragma unroll
    for (int ct = 0; ct < 4; ++ct) {
      v8f acc = zero_v8f();
      const _Float16* bW = wh + (size_t)C * C
                         + (size_t)(h * HD + ct * 16 + lc) * C + (half ? 16 : 0);
#pragma unroll 4
      for (int kk = 0; kk < KSTEPS; ++kk) {
        v8h a0 = *(const v8h*)&Xs[lc * 520 + kk * 32 + 16 * half];
        v8h a1 = *(const v8h*)&Xs[lc * 520 + kk * 32 + 16 * half + 8];
        v16h B = *(const v16h*)(bW + kk * 32);
        acc = wmma16(pack16(a0, a1), B, acc);
      }
      float bias = bk[h * HD + ct * 16 + lc];
      v8h kh;
#pragma unroll
      for (int r = 0; r < 8; ++r) {
        float v = acc[r] + bias;            // padded rows -> bias (K2 corrects)
        ssk        += v * v;
        ks_acc[ct] += v;
        kh[r] = (_Float16)v;
      }
      *(v8h*)&Kt[h * 1024 + ((ct * 16 + lc) << 4) + 8 * half] = kh;
    }

    // ---- V_h : GEMM + stage V^T ----
#pragma unroll
    for (int ct = 0; ct < 4; ++ct) {
      v8f acc = zero_v8f();
      const _Float16* bW = wh + (size_t)2 * C * C
                         + (size_t)(h * HD + ct * 16 + lc) * C + (half ? 16 : 0);
#pragma unroll 4
      for (int kk = 0; kk < KSTEPS; ++kk) {
        v8h a0 = *(const v8h*)&Xs[lc * 520 + kk * 32 + 16 * half];
        v8h a1 = *(const v8h*)&Xs[lc * 520 + kk * 32 + 16 * half + 8];
        v16h B = *(const v16h*)(bW + kk * 32);
        acc = wmma16(pack16(a0, a1), B, acc);
      }
      float bias = bv[h * HD + ct * 16 + lc];
      v8h vh;
#pragma unroll
      for (int r = 0; r < 8; ++r)
        vh[r] = (_Float16)(acc[r] + bias);
      *(v8h*)&Vt[h * 1024 + ((ct * 16 + lc) << 4) + 8 * half] = vh;
    }

    // ---- Q_h : GEMM, only for ||Q||^2 ----
#pragma unroll
    for (int ct = 0; ct < 4; ++ct) {
      v8f acc = zero_v8f();
      const _Float16* bW = wh
                         + (size_t)(h * HD + ct * 16 + lc) * C + (half ? 16 : 0);
#pragma unroll 4
      for (int kk = 0; kk < KSTEPS; ++kk) {
        v8h a0 = *(const v8h*)&Xs[lc * 520 + kk * 32 + 16 * half];
        v8h a1 = *(const v8h*)&Xs[lc * 520 + kk * 32 + 16 * half + 8];
        v16h B = *(const v16h*)(bW + kk * 32);
        acc = wmma16(pack16(a0, a1), B, acc);
      }
      float bias = bq[h * HD + ct * 16 + lc];
#pragma unroll
      for (int r = 0; r < 8; ++r) {
        float v = acc[r] + bias;
        ssq += v * v;
      }
    }

    // ---- ktv_h += K_h^T (64x16) x V_h (16x64), k padded to 32 ----
#pragma unroll
    for (int mt = 0; mt < 4; ++mt) {
      v8h ka = *(const v8h*)&Kt[h * 1024 + ((mt * 16 + lc) << 4) + 8 * half];
      v16h A = pack16(ka, zero_v8h());
#pragma unroll
      for (int dt = 0; dt < 4; ++dt) {
        v16h B;
        B = pack16(*(const v8h*)&Vt[h * 1024 + ((dt * 16 + lc) << 4) + 8 * half], zero_v8h());
        ktv_acc[mt * 4 + dt] = wmma16(A, B, ktv_acc[mt * 4 + dt]);
      }
    }
    __syncthreads();                        // done reading Xs for this tile
  }

  // ---- flush reductions (once per block) ----
#pragma unroll
  for (int ct = 0; ct < 4; ++ct) {
    float v = ks_acc[ct];
    v += __shfl_xor(v, 16, 32);
    if (lane < 16)
      atomicAdd(wsf + WS_KEYSUM / 4 + h * HD + ct * 16 + lc, v);
  }
#pragma unroll
  for (int mt = 0; mt < 4; ++mt)
#pragma unroll
    for (int dt = 0; dt < 4; ++dt)
#pragma unroll
      for (int r = 0; r < 8; ++r)
        atomicAdd(wsf + WS_KTV / 4
                      + (h * HD + mt * 16 + r + 8 * half) * HD + dt * 16 + lc,
                  ktv_acc[mt * 4 + dt][r]);
#pragma unroll
  for (int s = 16; s > 0; s >>= 1) {
    ssq += __shfl_xor(ssq, s, 32);
    ssk += __shfl_xor(ssk, s, 32);
  }
  if (lane == 0) { red[w] = ssq; red[8 + w] = ssk; }
  __syncthreads();
  if (threadIdx.x == 0) {
    float a = 0.f, b = 0.f;
#pragma unroll
    for (int i = 0; i < 8; ++i) { a += red[i]; b += red[8 + i]; }
    atomicAdd(wsf + 0, a);
    atomicAdd(wsf + 1, b);
  }
}

// =====================================================================
// K2: bias-correct the padded-row contributions (R = ntiles*16 - n),
//     then Mh[h][d][m] = ktv*rnormK (f16), kss = key_sum*rnormQ*rnormK.
// =====================================================================
__global__ void k2_finalize(float* __restrict__ wsf, _Float16* __restrict__ mh,
                            float* __restrict__ kss,
                            const float* __restrict__ bq,
                            const float* __restrict__ bk,
                            const float* __restrict__ bv, int n, int ntiles) {
  __shared__ float sh[2];
  const float fR = (float)(ntiles * TILE - n);
  if (threadIdx.x == 0) {
    float cq = 0.f, ck = 0.f;
    for (int c = 0; c < C; ++c) { cq += bq[c] * bq[c]; ck += bk[c] * bk[c]; }
    sh[0] = rsqrtf(wsf[0] - fR * cq);
    sh[1] = rsqrtf(wsf[1] - fR * ck);
  }
  __syncthreads();
  float rnq = sh[0], rnk = sh[1];
  int gid = blockIdx.x * blockDim.x + threadIdx.x;
  if (gid == 0) wsf[2] = rnq;
  const float* ktv  = wsf + WS_KTV / 4;
  const float* ksum = wsf + WS_KEYSUM / 4;
  for (int i = gid; i < H * HD * HD; i += gridDim.x * blockDim.x) {
    int hh = i >> 12, m = (i >> 6) & 63, d = i & 63;
    float corr = fR * bk[hh * HD + m] * bv[hh * HD + d];
    mh[(hh << 12) + (d << 6) + m] = (_Float16)((ktv[i] - corr) * rnk);
  }
  if (gid < C) kss[gid] = (ksum[gid] - fR * bk[gid]) * rnq * rnk;
}

// =====================================================================
// K3: recompute Q_h and V_h per 16-node tile, fuse epilogue:
//     out = (Q@Mh * rnormQ + n*V) / (Q . kss + n)
// =====================================================================
__global__ __launch_bounds__(256) void
k3_output(const float* __restrict__ x,
          const float* __restrict__ bq, const float* __restrict__ bv,
          const float* __restrict__ wsf, const _Float16* __restrict__ wh,
          const _Float16* __restrict__ mh, const float* __restrict__ kss,
          float* __restrict__ out, int n) {
  extern __shared__ char smem[];
  float*    Xs32 = (float*)(smem + L1_XS32);
  _Float16* Xs   = (_Float16*)(smem + L1_XS);
  _Float16* Qs   = (_Float16*)(smem + L3_QS);
  float*    den  = (float*)(smem + L3_DEN);

  const int w    = threadIdx.x >> 5;
  const int lane = threadIdx.x & 31;
  const int half = lane >> 4;
  const int lc   = lane & 15;
  const int h    = w;
  const int nb   = blockIdx.x * TILE;

#if HAVE_TDM
  if (w == 0)
    tdm_load_x(x + (size_t)nb * C, n - nb,
               (unsigned)(unsigned long long)(size_t)(void*)Xs32);
  if (lane < 16) den[w * 16 + lane] = 0.f;
  if (w == 0) __builtin_amdgcn_s_wait_tensorcnt(0);
  __syncthreads();
  for (int i = threadIdx.x; i < TILE * C / 4; i += 256) {
    int row = i >> 7;
    int c4  = (i & 127) * 4;
    float4 xv = *(const float4*)&Xs32[row * C + c4];
    Xs[row * 520 + c4 + 0] = (_Float16)xv.x;
    Xs[row * 520 + c4 + 1] = (_Float16)xv.y;
    Xs[row * 520 + c4 + 2] = (_Float16)xv.z;
    Xs[row * 520 + c4 + 3] = (_Float16)xv.w;
  }
#else
  for (int i = threadIdx.x; i < TILE * C / 4; i += 256) {
    int row = i >> 7;
    int c4  = (i & 127) * 4;
    int node = nb + row;
    float4 xv = make_float4(0.f, 0.f, 0.f, 0.f);
    if (node < n) xv = *(const float4*)(x + (size_t)node * C + c4);
    Xs[row * 520 + c4 + 0] = (_Float16)xv.x;
    Xs[row * 520 + c4 + 1] = (_Float16)xv.y;
    Xs[row * 520 + c4 + 2] = (_Float16)xv.z;
    Xs[row * 520 + c4 + 3] = (_Float16)xv.w;
  }
  if (lane < 16) den[w * 16 + lane] = 0.f;
#endif
  __syncthreads();

  // ---- Q_h GEMM -> stage A-fragments + LDS-atomic denominator ----
#pragma unroll
  for (int ct = 0; ct < 4; ++ct) {
    v8f acc = zero_v8f();
    const _Float16* bW = wh + (size_t)(h * HD + ct * 16 + lc) * C + (half ? 16 : 0);
#pragma unroll 4
    for (int kk = 0; kk < KSTEPS; ++kk) {
      v8h a0 = *(const v8h*)&Xs[lc * 520 + kk * 32 + 16 * half];
      v8h a1 = *(const v8h*)&Xs[lc * 520 + kk * 32 + 16 * half + 8];
      v16h B = *(const v16h*)(bW + kk * 32);
      acc = wmma16(pack16(a0, a1), B, acc);
    }
    float bias = bq[h * HD + ct * 16 + lc];
    float ksv  = kss[h * HD + ct * 16 + lc];
#pragma unroll
    for (int r = 0; r < 8; ++r) {
      float v = acc[r] + bias;
      atomicAdd(&den[w * 16 + r + 8 * half], v * ksv);
      Qs[w * 1280 + (r + 8 * half) * 80 + ct * 16 + lc] = (_Float16)v;
    }
  }
  __syncthreads();

  const float rnq = wsf[2];
  const float fn  = (float)n;
  v16h QA[2];
#pragma unroll
  for (int kk2 = 0; kk2 < 2; ++kk2) {
    v8h a0 = *(const v8h*)&Qs[w * 1280 + lc * 80 + kk2 * 32 + 16 * half];
    v8h a1 = *(const v8h*)&Qs[w * 1280 + lc * 80 + kk2 * 32 + 16 * half + 8];
    QA[kk2] = pack16(a0, a1);
  }
  float rden[8];
#pragma unroll
  for (int r = 0; r < 8; ++r)
    rden[r] = 1.0f / (den[w * 16 + r + 8 * half] + fn);

  // ---- per d-tile: V_h GEMM + num = Q@Mh + n*V, divide, store ----
#pragma unroll
  for (int dt = 0; dt < 4; ++dt) {
    v8f vacc = zero_v8f();
    const _Float16* bW = wh + (size_t)2 * C * C
                       + (size_t)(h * HD + dt * 16 + lc) * C + (half ? 16 : 0);
#pragma unroll 4
    for (int kk = 0; kk < KSTEPS; ++kk) {
      v8h a0 = *(const v8h*)&Xs[lc * 520 + kk * 32 + 16 * half];
      v8h a1 = *(const v8h*)&Xs[lc * 520 + kk * 32 + 16 * half + 8];
      v16h B = *(const v16h*)(bW + kk * 32);
      vacc = wmma16(pack16(a0, a1), B, vacc);
    }
    v8f num = zero_v8f();
#pragma unroll
    for (int kk2 = 0; kk2 < 2; ++kk2) {
      v16h B = *(const v16h*)(mh + (h << 12) + ((dt * 16 + lc) << 6)
                                 + kk2 * 32 + 16 * half);
      num = wmma16(QA[kk2], B, num);
    }
    float biasv = bv[h * HD + dt * 16 + lc];
#pragma unroll
    for (int r = 0; r < 8; ++r) {
      int node = nb + r + 8 * half;
      if (node < n) {
        float vv = vacc[r] + biasv;
        float o  = (num[r] * rnq + fn * vv) * rden[r];
        out[(size_t)node * C + h * HD + dt * 16 + lc] = o;
      }
    }
  }
}

// =====================================================================
extern "C" void kernel_launch(void* const* d_in, const int* in_sizes, int n_in,
                              void* d_out, int out_size, void* d_ws, size_t ws_size,
                              hipStream_t stream) {
  (void)n_in; (void)out_size; (void)ws_size;
  const float* x  = (const float*)d_in[0];
  const float* wq = (const float*)d_in[1];
  const float* bq = (const float*)d_in[2];
  const float* wk = (const float*)d_in[3];
  const float* bk = (const float*)d_in[4];
  const float* wv = (const float*)d_in[5];
  const float* bv = (const float*)d_in[6];
  // d_in[7] = edge_index: unused by the reference math.

  const int n      = in_sizes[0] / C;
  const int ntiles = (n + TILE - 1) / TILE;

  float*     wsf = (float*)d_ws;
  _Float16*  wh  = (_Float16*)((char*)d_ws + WS_WH);
  _Float16*  mh  = (_Float16*)((char*)d_ws + WS_MH);
  float*     kss = (float*)((char*)d_ws + WS_KSS);
  float*     out = (float*)d_out;

  zero_red<<<dim3(132), dim3(256), 0, stream>>>(wsf);
  cvt_w<<<dim3(3072), dim3(256), 0, stream>>>(wq, wk, wv, wh);

  int nb1 = 640;
  if (nb1 > ntiles) nb1 = ntiles;
  k1_reduce<<<dim3(nb1), dim3(256), (size_t)L1_BYTES, stream>>>(
      x, bq, bk, bv, wsf, wh, n, ntiles);
  k2_finalize<<<dim3(32), dim3(256), 0, stream>>>(wsf, mh, kss, bq, bk, bv, n, ntiles);
  k3_output<<<dim3(ntiles), dim3(256), (size_t)L3_BYTES, stream>>>(
      x, bq, bv, wsf, wh, mh, kss, out, n);
}
